// GRUModel_85529978733196
// MI455X (gfx1250) — matrix operand order repaired and checked by reading, not compiled, not run on hardware
//
#include <hip/hip_runtime.h>
#include <hip/hip_bf16.h>
#include <math.h>

#define B_   4096
#define H_   512
#define G_   1536   // 3*H
#define V_   512
#define T_   31
#define DIN_ 14
#define NK_  (H_ / 32)   // k-steps per GEMM

typedef __attribute__((ext_vector_type(16))) _Float16 v16h;
typedef __attribute__((ext_vector_type(8)))  _Float16 v8h;
typedef __attribute__((ext_vector_type(8)))  float    v8f;
typedef int v4i __attribute__((vector_size(16)));   // matches async builtin pointee

#if __has_builtin(__builtin_amdgcn_global_load_async_to_lds_b128) && \
    __has_builtin(__builtin_amdgcn_s_wait_asynccnt)
#define HAVE_ASYNC_LDS 1
#endif

typedef __attribute__((address_space(1))) v4i* gvec_p;   // global (AS1)
typedef __attribute__((address_space(3))) v4i* lvec_p;   // LDS (AS3)

// ---- async global->LDS staging (16B per call) --------------------------
__device__ __forceinline__ void stage16(const _Float16* g, _Float16* l) {
#ifdef HAVE_ASYNC_LDS
  __builtin_amdgcn_global_load_async_to_lds_b128(
      (gvec_p)(__attribute__((address_space(1))) void*)(void*)g,
      (lvec_p)(__attribute__((address_space(3))) void*)(void*)l, 0, 0);
#else
  *(v8h*)l = *(const v8h*)g;
#endif
}

// wait until at most `allow` async stages outstanding, then block barrier
__device__ __forceinline__ void stage_wait_keep1() {   // pipelined: older tile done
#ifdef HAVE_ASYNC_LDS
  __builtin_amdgcn_s_wait_asynccnt(1);
#endif
  __syncthreads();
}
__device__ __forceinline__ void stage_wait_all() {     // drain
#ifdef HAVE_ASYNC_LDS
  __builtin_amdgcn_s_wait_asynccnt(0);
#endif
  __syncthreads();
}

// ---- WMMA fragment helpers ---------------------------------------------
// A fragment (16x32 f16) from LDS tile, row-major, `stride` halves.
// lane<16: row=lane, K=0..15 ; lane>=16: row=lane-16, K=16..31
__device__ __forceinline__ v16h a_frag_lds(const _Float16* tile, int stride) {
  const int lane = threadIdx.x & 31;
  const _Float16* p = tile + (lane & 15) * stride + ((lane >> 4) ? 16 : 0);
  v8h lo = *(const v8h*)p;
  v8h hi = *(const v8h*)(p + 8);
  v16h r;
#pragma unroll
  for (int i = 0; i < 8; ++i) { r[i] = lo[i]; r[8 + i] = hi[i]; }
  return r;
}

// B fragment (32x16 f16) = W^T tile straight from row-major W[N,K].
__device__ __forceinline__ v16h b_frag_g(const _Float16* W, int jbase, int k, int stride) {
  const int lane = threadIdx.x & 31;
  const _Float16* p = W + (size_t)(jbase + (lane & 15)) * stride + k + ((lane >> 4) << 4);
  return *(const v16h*)p;  // contiguous 32B per lane
}

__device__ __forceinline__ v8f wmma32(v16h a, v16h b, v8f c) {
  return __builtin_amdgcn_wmma_f32_16x16x32_f16(false, a, false, b, (short)0, c,
                                                false, false);
}

__device__ __forceinline__ float sigmoidf_(float x) {
  return 1.0f / (1.0f + __expf(-x));
}

// ---- small prep kernels -------------------------------------------------
__global__ void __launch_bounds__(256) f32_to_f16_kernel(const float* __restrict__ s,
                                                         _Float16* __restrict__ d, int n) {
  int i = blockIdx.x * 256 + threadIdx.x;
  if (i < n) d[i] = (_Float16)s[i];
}

__global__ void __launch_bounds__(256) zero_f16_kernel(_Float16* __restrict__ p, int n) {
  int i = blockIdx.x * 256 + threadIdx.x;
  if (i < n) p[i] = (_Float16)0.0f;
}

__global__ void __launch_bounds__(256) inproj_kernel(const float* __restrict__ inp,
                                                     const float* __restrict__ Win,
                                                     const float* __restrict__ b_in,
                                                     _Float16* __restrict__ xh) {
  int idx = blockIdx.x * 256 + threadIdx.x;  // b*H + h
  if (idx >= B_ * H_) return;
  int b = idx >> 9, h = idx & (H_ - 1);
  float s = b_in[h];
#pragma unroll
  for (int d = 0; d < DIN_; ++d) s += inp[b * DIN_ + d] * Win[h * DIN_ + d];
  xh[idx] = (_Float16)s;
}

// ---- gi = x @ W_ih^T + b_ih  (M=4096, N=1536, K=512) --------------------
// block: 64 batch x 64 cols, 8 waves; wave = 32x16; double-buffered LDS A
__global__ void __launch_bounds__(256) gi_gemm_kernel(const _Float16* __restrict__ xh,
                                                      const _Float16* __restrict__ Wih,
                                                      const float* __restrict__ b_ih,
                                                      float* __restrict__ gi) {
  __shared__ _Float16 As[2][64][48];
  const int wave = threadIdx.x >> 5, lane = threadIdx.x & 31;
  const int bm = blockIdx.x * 64, jn = blockIdx.y * 64;
  const int wm = (wave & 1) * 32, wn = (wave >> 1) * 16;
  v8f acc0 = {}, acc1 = {};
  const int sr = threadIdx.x >> 2;       // 0..63
  const int sc = (threadIdx.x & 3) * 8;  // 0,8,16,24 halves
  const _Float16* arow = xh + (size_t)(bm + sr) * H_ + sc;

  stage16(arow, &As[0][sr][sc]);  // prologue: tile 0
  for (int i = 0; i < NK_; ++i) {
    const int k = i * 32;
    if (i + 1 < NK_) {
      stage16(arow + k + 32, &As[(i + 1) & 1][sr][sc]);
      stage_wait_keep1();           // tile i landed; tile i+1 in flight
    } else {
      stage_wait_all();
    }
    v16h a0 = a_frag_lds(&As[i & 1][wm][0], 48);
    v16h a1 = a_frag_lds(&As[i & 1][wm + 16][0], 48);
    v16h b  = b_frag_g(Wih, jn + wn, k, H_);
    acc0 = wmma32(a0, b, acc0);
    acc1 = wmma32(a1, b, acc1);
    __syncthreads();                // done reading tile i before it is restaged
  }
  const int col = lane & 15, rb = (lane >> 4) * 8;
  const int j = jn + wn + col;
  const float bj = b_ih[j];
#pragma unroll
  for (int r = 0; r < 8; ++r) {
    gi[(size_t)(bm + wm + rb + r) * G_ + j]      = acc0[r] + bj;
    gi[(size_t)(bm + wm + 16 + rb + r) * G_ + j] = acc1[r] + bj;
  }
}

// ---- fused GRU step: gh = h @ W_hh^T (3 gates), gate math ---------------
// block: 64 batch x 64 j-cols, 8 waves; wave = 32x16 x 3 gates (6 accums)
__global__ void __launch_bounds__(256) gru_step_kernel(
    const _Float16* __restrict__ h16prev, _Float16* __restrict__ h16next,
    const _Float16* __restrict__ Whh, const float* __restrict__ b_hh,
    const float* __restrict__ gi, const float* __restrict__ hprev32,
    float* __restrict__ hout32) {
  __shared__ _Float16 As[2][64][48];
  const int wave = threadIdx.x >> 5, lane = threadIdx.x & 31;
  const int bm = blockIdx.x * 64, jn = blockIdx.y * 64;
  const int wm = (wave & 1) * 32, wn = (wave >> 1) * 16;
  v8f acc[3][2] = {};
  const int sr = threadIdx.x >> 2, sc = (threadIdx.x & 3) * 8;
  const _Float16* arow = h16prev + (size_t)(bm + sr) * H_ + sc;

  stage16(arow, &As[0][sr][sc]);
  for (int i = 0; i < NK_; ++i) {
    const int k = i * 32;
    if (i + 1 < NK_) {
      stage16(arow + k + 32, &As[(i + 1) & 1][sr][sc]);
      __builtin_prefetch(Whh + (size_t)(jn + wn + (lane & 15)) * H_ + k + 32, 0, 3);
      stage_wait_keep1();
    } else {
      stage_wait_all();
    }
    // 3 unique B fragments (one per gate), 2 A fragments (two 16-row tiles)
    v16h b0 = b_frag_g(Whh, 0 * H_ + jn + wn, k, H_);
    v16h b1 = b_frag_g(Whh, 1 * H_ + jn + wn, k, H_);
    v16h b2 = b_frag_g(Whh, 2 * H_ + jn + wn, k, H_);
    v16h a0 = a_frag_lds(&As[i & 1][wm][0], 48);
    v16h a1 = a_frag_lds(&As[i & 1][wm + 16][0], 48);
    acc[0][0] = wmma32(a0, b0, acc[0][0]);
    acc[1][0] = wmma32(a0, b1, acc[1][0]);
    acc[2][0] = wmma32(a0, b2, acc[2][0]);
    acc[0][1] = wmma32(a1, b0, acc[0][1]);
    acc[1][1] = wmma32(a1, b1, acc[1][1]);
    acc[2][1] = wmma32(a1, b2, acc[2][1]);
    __syncthreads();
  }
  const int col = lane & 15, rb = (lane >> 4) * 8;
  const int j = jn + wn + col;
  const float br = b_hh[j], bz = b_hh[H_ + j], bn = b_hh[2 * H_ + j];
#pragma unroll
  for (int m = 0; m < 2; ++m) {
#pragma unroll
    for (int r = 0; r < 8; ++r) {
      const int brow = bm + wm + m * 16 + rb + r;
      const size_t gb = (size_t)brow * G_;
      const float rg = sigmoidf_(gi[gb + j] + acc[0][m][r] + br);
      const float zg = sigmoidf_(gi[gb + H_ + j] + acc[1][m][r] + bz);
      const float ng = tanhf(gi[gb + 2 * H_ + j] + rg * (acc[2][m][r] + bn));
      const float hp = hprev32 ? hprev32[(size_t)brow * (T_ * H_) + j] : 0.0f;
      const float hn = (1.0f - zg) * ng + zg * hp;
      hout32[(size_t)brow * (T_ * H_) + j] = hn;            // = hidden_flat output
      h16next[(size_t)brow * H_ + j] = (_Float16)hn;        // next-step WMMA operand
    }
  }
}

// ---- logits = hs @ W_out^T + b_out, softmax over V ----------------------
// block: 16 (b,t) rows x full V=512; 8 waves, each 16x64 (4 accums)
__global__ void __launch_bounds__(256) out_kernel(const float* __restrict__ hidden,
                                                  const _Float16* __restrict__ Wout,
                                                  const float* __restrict__ b_out,
                                                  float* __restrict__ out) {
  __shared__ _Float16 As[16][528];
  __shared__ float Ls[16][516];
  const size_t row0 = (size_t)blockIdx.x * 16;
  for (int i = threadIdx.x; i < 16 * H_; i += 256) {
    int r = i >> 9, c = i & (H_ - 1);
    As[r][c] = (_Float16)hidden[(row0 + r) * H_ + c];
  }
  __syncthreads();
  const int wave = threadIdx.x >> 5, lane = threadIdx.x & 31;
  const int n0 = wave * 64;
  v8f acc[4] = {};
  for (int k = 0; k < H_; k += 32) {
    v16h a = a_frag_lds(&As[0][k], 528);
#pragma unroll
    for (int nt = 0; nt < 4; ++nt) {
      v16h b = b_frag_g(Wout, n0 + nt * 16, k, H_);
      acc[nt] = wmma32(a, b, acc[nt]);
    }
  }
  const int col = lane & 15, rb = (lane >> 4) * 8;
#pragma unroll
  for (int nt = 0; nt < 4; ++nt) {
    const int c = n0 + nt * 16 + col;
    const float bc = b_out[c];
#pragma unroll
    for (int r = 0; r < 8; ++r) Ls[rb + r][c] = acc[nt][r] + bc;
  }
  __syncthreads();
  // softmax: wave w owns rows 2w, 2w+1; lane-strided over 512 cols
#pragma unroll
  for (int rr = 0; rr < 2; ++rr) {
    const int r = wave * 2 + rr;
    float mx = -INFINITY;
    for (int c = lane; c < V_; c += 32) mx = fmaxf(mx, Ls[r][c]);
    for (int o = 16; o; o >>= 1) mx = fmaxf(mx, __shfl_xor(mx, o, 32));
    float s = 0.0f;
    for (int c = lane; c < V_; c += 32) s += __expf(Ls[r][c] - mx);
    for (int o = 16; o; o >>= 1) s += __shfl_xor(s, o, 32);
    const float inv = 1.0f / s;
    float* orow = out + (row0 + r) * V_;
    for (int c = lane; c < V_; c += 32)
      __builtin_nontemporal_store(__expf(Ls[r][c] - mx) * inv, orow + c);  // stream 260MB past L2
  }
}

// ---- host orchestration -------------------------------------------------
extern "C" void kernel_launch(void* const* d_in, const int* in_sizes, int n_in,
                              void* d_out, int out_size, void* d_ws, size_t ws_size,
                              hipStream_t stream) {
  (void)in_sizes; (void)n_in; (void)out_size; (void)ws_size;
  const float* inputs = (const float*)d_in[0];
  const float* W_in   = (const float*)d_in[1];
  const float* b_in   = (const float*)d_in[2];
  const float* W_ih   = (const float*)d_in[3];
  const float* b_ih   = (const float*)d_in[4];
  const float* W_hh   = (const float*)d_in[5];
  const float* b_hh   = (const float*)d_in[6];
  const float* W_out  = (const float*)d_in[7];
  const float* b_out  = (const float*)d_in[8];

  char* ws = (char*)d_ws;
  _Float16* Wih16  = (_Float16*)ws; ws += (size_t)G_ * H_ * 2;
  _Float16* Whh16  = (_Float16*)ws; ws += (size_t)G_ * H_ * 2;
  _Float16* Wout16 = (_Float16*)ws; ws += (size_t)V_ * H_ * 2;
  _Float16* xh     = (_Float16*)ws; ws += (size_t)B_ * H_ * 2;
  _Float16* h16a   = (_Float16*)ws; ws += (size_t)B_ * H_ * 2;
  _Float16* h16b   = (_Float16*)ws; ws += (size_t)B_ * H_ * 2;
  float*    gi     = (float*)ws;    // B_*G_ fp32

  float* outp   = (float*)d_out;                         // [B*T, V]
  float* hidden = (float*)d_out + (size_t)B_ * T_ * V_;  // [B][T][H] == hidden_flat

  f32_to_f16_kernel<<<(G_ * H_ + 255) / 256, 256, 0, stream>>>(W_ih, Wih16, G_ * H_);
  f32_to_f16_kernel<<<(G_ * H_ + 255) / 256, 256, 0, stream>>>(W_hh, Whh16, G_ * H_);
  f32_to_f16_kernel<<<(V_ * H_ + 255) / 256, 256, 0, stream>>>(W_out, Wout16, V_ * H_);
  zero_f16_kernel<<<(B_ * H_ + 255) / 256, 256, 0, stream>>>(h16a, B_ * H_);
  inproj_kernel<<<(B_ * H_ + 255) / 256, 256, 0, stream>>>(inputs, W_in, b_in, xh);

  gi_gemm_kernel<<<dim3(B_ / 64, G_ / 64), 256, 0, stream>>>(xh, Wih16, b_ih, gi);

  _Float16* hb[2] = {h16a, h16b};
  for (int t = 0; t < T_; ++t) {
    const float* hprev32 = (t == 0) ? nullptr : hidden + (size_t)(t - 1) * H_;
    gru_step_kernel<<<dim3(B_ / 64, H_ / 64), 256, 0, stream>>>(
        hb[t & 1], hb[(t + 1) & 1], Whh16, b_hh, gi, hprev32,
        hidden + (size_t)t * H_);
  }

  out_kernel<<<(B_ * T_) / 16, 256, 0, stream>>>(hidden, Wout16, b_out, outp);
}
